// LSTM_77386720739803
// MI455X (gfx1250) — matrix operand order repaired and checked by reading, not compiled, not run on hardware
//
#include <hip/hip_runtime.h>
#include <stdint.h>

// Problem constants (match reference)
#define B_ 128
#define S_ 128
#define E_ 512
#define H_ 1024
#define G_ 4096   // 4*H
#define V_ 10000

typedef __attribute__((ext_vector_type(16))) __bf16    v16bf;
typedef __attribute__((ext_vector_type(8)))  __bf16    v8bf;
typedef __attribute__((ext_vector_type(8)))  float     v8f;
typedef __attribute__((ext_vector_type(4)))  uint32_t  u32x4;
typedef __attribute__((ext_vector_type(8)))  int32_t   i32x8;
typedef __attribute__((ext_vector_type(4)))  int32_t   i32x4;

union V16U { v16bf v; v8bf h[2]; };

__device__ __forceinline__ uint16_t f2bf(float f) {
  union { float f; uint32_t u; } x; x.f = f;
  uint32_t r = x.u + 0x7FFFu + ((x.u >> 16) & 1u);   // round-to-nearest-even
  return (uint16_t)(r >> 16);
}
__device__ __forceinline__ float bf2f(uint16_t b) {
  union { uint32_t u; float f; } x; x.u = ((uint32_t)b) << 16;
  return x.f;
}

// ---------------------------------------------------------------------------
// WMMA fragment helpers (wave32; contiguous k order used for both A and B, not the ISA table order):
//  A (16xK row-major, bf16): lane holds row m = lane%16; lanes 0-15 take K
//    {k0..k0+15} contiguous, lanes 16-31 take {k0+16..k0+31}.
//  B passed transposed (N,K) row-major: lane holds column n = lane%16;
//    lanes 0-15 take K {k0..k0+15} contiguous, lanes 16-31 take {k0+16..k0+31}.
// ---------------------------------------------------------------------------
__device__ __forceinline__ v16bf load_frag_a(const uint16_t* __restrict__ row,
                                             int k0, int aoff) {
  V16U a;
  a.h[0] = *(const v8bf*)(row + k0 + aoff);
  a.h[1] = *(const v8bf*)(row + k0 + aoff + 8);
  return a.v;
}
__device__ __forceinline__ v16bf load_frag_b(const uint16_t* __restrict__ row,
                                             int k0, int boff) {
  const v8bf* pb = (const v8bf*)(row + k0 + boff);
  V16U b;
  b.h[0] = pb[0];
  b.h[1] = pb[1];
  return b.v;
}
__device__ __forceinline__ v8f wmma_bf16(v16bf a, v16bf b, v8f c) {
  return __builtin_amdgcn_wmma_f32_16x16x32_bf16(false, a, false, b,
                                                 (short)0, c, false, false);
}

// Register-direct 2x2-tile K-loop: 4 WMMAs per fragment-load set (A and B
// fragments each reused twice -> 16 FLOP/byte of fragment traffic).
template <int K>
__device__ __forceinline__ void bf16_gemm_2x2(
    const uint16_t* __restrict__ ar0, const uint16_t* __restrict__ ar1,
    const uint16_t* __restrict__ br0, const uint16_t* __restrict__ br1,
    v8f& c00, v8f& c01, v8f& c10, v8f& c11) {
  const int lane = threadIdx.x & 31;
  const int aoff = (lane & 16) ? 16 : 0;
  const int boff = (lane & 16) ? 16 : 0;
#pragma unroll 2
  for (int k0 = 0; k0 < K; k0 += 32) {
    __builtin_prefetch(br0 + k0 + 256, 0, 0);  // global_prefetch_b8
    __builtin_prefetch(br1 + k0 + 256, 0, 0);
    v16bf a0 = load_frag_a(ar0, k0, aoff);
    v16bf a1 = load_frag_a(ar1, k0, aoff);
    v16bf b0 = load_frag_b(br0, k0, boff);
    v16bf b1 = load_frag_b(br1, k0, boff);
    c00 = wmma_bf16(a0, b0, c00);
    c01 = wmma_bf16(a0, b1, c01);
    c10 = wmma_bf16(a1, b0, c10);
    c11 = wmma_bf16(a1, b1, c11);
  }
}

// ---------------------------------------------------------------------------
// Tensor Data Mover: DMA a 2D tile (tile_dim1 lines x tile_dim0 elems, bf16)
// from a row-major tensor into LDS, inserting 4 DWORDs of pad per 32 DWORDs
// (64 bf16) so LDS line stride is 72 elements = 144 B (16B aligned, distinct
// bank groups per line).  D# packing per CDNA5 ISA.
//
// NOTE: hipcc compiles this TU twice.  The TDM builtin only exists in the
// DEVICE pass, so assert availability there and stub the host pass (host
// never executes __device__ bodies).
// ---------------------------------------------------------------------------
#if defined(__HIP_DEVICE_COMPILE__)
  #if __has_builtin(__builtin_amdgcn_tensor_load_to_lds)
    #define HAVE_TDM 1
  #else
    #error "gfx1250 TDM builtin __builtin_amdgcn_tensor_load_to_lds missing in device pass"
  #endif
#else
  #define HAVE_TDM 0   // host pass: parse-only stub
#endif

__device__ __forceinline__ void tdm_load_2d_bf16(
    uint32_t lds_byte_off, const void* tile_start,
    uint32_t tensor_dim0, uint32_t tensor_dim1, uint64_t dim0_stride_elems,
    uint32_t tile_dim0, uint32_t tile_dim1) {
#if HAVE_TDM
  uint64_t ga = (uint64_t)(uintptr_t)tile_start;
  u32x4 g0;
  g0[0] = 1u;                                   // count=1, user mode
  g0[1] = lds_byte_off;                         // LDS dest (bytes)
  g0[2] = (uint32_t)ga;                         // global_addr[31:0]
  g0[3] = (uint32_t)((ga >> 32) & 0x01FFFFFFu)  // global_addr[56:32]
          | (2u << 30);                         // type = 2 ("image")

  const uint32_t data_size    = 1;  // 2-byte elements
  const uint32_t pad_enable   = 1;
  const uint32_t pad_interval = 4;  // 2^(4+1) = 32 DWORDs (64 bf16) per line
  const uint32_t pad_amount   = 3;  // 3+1 = 4 DWORDs (8 bf16) of pad
  i32x8 g1;
  g1[0] = (int32_t)((data_size << 16) | (pad_enable << 20) |
                    (pad_interval << 22) | (pad_amount << 25));
  g1[1] = (int32_t)((tensor_dim0 & 0xFFFFu) << 16);            // tdim0[15:0]
  g1[2] = (int32_t)((tensor_dim0 >> 16) |                      // tdim0[31:16]
                    ((tensor_dim1 & 0xFFFFu) << 16));          // tdim1[15:0]
  g1[3] = (int32_t)((tensor_dim1 >> 16) | (tile_dim0 << 16));  // tdim1 | tile0
  g1[4] = (int32_t)(tile_dim1 & 0xFFFFu);                      // tile1, tile2=0
  g1[5] = (int32_t)(uint32_t)(dim0_stride_elems & 0xFFFFFFFFu);
  g1[6] = (int32_t)(uint32_t)((dim0_stride_elems >> 32) & 0xFFFFu);  // stride1=0
  g1[7] = 0;
  i32x4 z4 = {0, 0, 0, 0};
#if defined(__clang_major__) && __clang_major__ >= 23
  i32x8 z8 = {0, 0, 0, 0, 0, 0, 0, 0};
  __builtin_amdgcn_tensor_load_to_lds(g0, g1, z4, z4, z8, 0);
#else
  __builtin_amdgcn_tensor_load_to_lds(g0, g1, z4, z4, 0);
#endif
#else
  (void)lds_byte_off; (void)tile_start; (void)tensor_dim0; (void)tensor_dim1;
  (void)dim0_stride_elems; (void)tile_dim0; (void)tile_dim1;
#endif
}

// ---------------------------------------------------------------------------
// Prep kernels: fp32 -> bf16 conversions (once per call, amortized)
// ---------------------------------------------------------------------------
__global__ void k_cvt(const float* __restrict__ src, uint16_t* __restrict__ dst, int n) {
  int i = blockIdx.x * blockDim.x + threadIdx.x;
  int stride = gridDim.x * blockDim.x;
  for (; i < n; i += stride) dst[i] = f2bf(src[i]);
}

// (R,C) fp32 row-major -> (C,R) bf16 row-major (store B transposed)
__global__ void k_cvt_t(const float* __restrict__ src, uint16_t* __restrict__ dst,
                        int R, int C) {
  int i = blockIdx.x * blockDim.x + threadIdx.x;
  int n = R * C;
  int stride = gridDim.x * blockDim.x;
  for (; i < n; i += stride) {
    int r = i / C, c = i - r * C;
    dst[(size_t)c * R + r] = f2bf(src[i]);
  }
}

__global__ void k_zero(float* __restrict__ p, int n) {
  int i = blockIdx.x * blockDim.x + threadIdx.x;
  if (i < n) p[i] = 0.0f;
}

// ---------------------------------------------------------------------------
// Phase 1: fused embedding gather + xW = x@W + bias  -> xwB (B*S, 4H) bf16
// One 32x32 output tile (2x2 WMMA accs) per wave; 8 waves/block along N.
// ---------------------------------------------------------------------------
__global__ void k_embed_gemm(const int* __restrict__ inp,        // (B*S)
                             const uint16_t* __restrict__ embB,  // (V,E) bf16
                             const uint16_t* __restrict__ wT,    // (4H,E) bf16 (W^T)
                             const float* __restrict__ bias,     // (4H)
                             uint16_t* __restrict__ xwB) {       // (B*S,4H) bf16
  const int wave = threadIdx.x >> 5;
  const int lane = threadIdx.x & 31;
  const int l15  = lane & 15;
  const int m32  = blockIdx.x * 32;                 // M/32 = 512
  const int n32  = (blockIdx.y * 8 + wave) * 32;    // 4H/32 = 128

  const int tok0 = inp[m32 + l15];                  // gather for both m tiles
  const int tok1 = inp[m32 + 16 + l15];
  const uint16_t* ar0 = embB + (size_t)tok0 * E_;
  const uint16_t* ar1 = embB + (size_t)tok1 * E_;
  const uint16_t* br0 = wT + (size_t)(n32 + l15) * E_;
  const uint16_t* br1 = wT + (size_t)(n32 + 16 + l15) * E_;

  v8f c00 = {}, c01 = {}, c10 = {}, c11 = {};
  bf16_gemm_2x2<E_>(ar0, ar1, br0, br1, c00, c01, c10, c11);

  const int mrow = m32 + ((lane & 16) ? 8 : 0);
  const int nc0 = n32 + l15, nc1 = nc0 + 16;
  const float bv0 = bias[nc0], bv1 = bias[nc1];
#pragma unroll
  for (int i = 0; i < 8; ++i) {
    xwB[(size_t)(mrow + i) * G_ + nc0]      = f2bf(c00[i] + bv0);
    xwB[(size_t)(mrow + i) * G_ + nc1]      = f2bf(c01[i] + bv1);
    xwB[(size_t)(mrow + 16 + i) * G_ + nc0] = f2bf(c10[i] + bv0);
    xwB[(size_t)(mrow + 16 + i) * G_ + nc1] = f2bf(c11[i] + bv1);
  }
}

// ---------------------------------------------------------------------------
// Phase 2a (per timestep): g = xw[:,t,:] + h_{t-1}@U, apply gate nonlinearity.
// 2x2 WMMA tiles per wave: halves U fragment traffic on the serial chain.
// ---------------------------------------------------------------------------
__global__ void k_lstm_gates(const uint16_t* __restrict__ hprev, // bf16 or null
                             const uint16_t* __restrict__ uT,    // (4H,H) bf16 (U^T)
                             const uint16_t* __restrict__ xwB,   // (B*S,4H) bf16
                             float* __restrict__ gact,           // (B,4H) f32
                             int t) {
  const int wave = threadIdx.x >> 5;
  const int lane = threadIdx.x & 31;
  const int l15  = lane & 15;
  const int m32  = blockIdx.x * 32;                 // B/32 = 4
  const int n32  = (blockIdx.y * 8 + wave) * 32;    // 4H/32 = 128

  v8f c00 = {}, c01 = {}, c10 = {}, c11 = {};
  if (hprev != nullptr) {
    const uint16_t* ar0 = hprev + (size_t)(m32 + l15) * (S_ * H_);
    const uint16_t* ar1 = hprev + (size_t)(m32 + 16 + l15) * (S_ * H_);
    const uint16_t* br0 = uT + (size_t)(n32 + l15) * H_;
    const uint16_t* br1 = uT + (size_t)(n32 + 16 + l15) * H_;
    bf16_gemm_2x2<H_>(ar0, ar1, br0, br1, c00, c01, c10, c11);
  }

  const int mrow = m32 + ((lane & 16) ? 8 : 0);
  const int nc0 = n32 + l15, nc1 = nc0 + 16;
  const int g0t = nc0 >> 10, g1t = nc1 >> 10;       // gate id (0=i 1=f 2=g 3=o)
#pragma unroll
  for (int i = 0; i < 8; ++i) {
    const int ma = mrow + i, mb = mrow + 16 + i;
    float ga = c00[i] + bf2f(xwB[((size_t)ma * S_ + t) * G_ + nc0]);
    float gb = c01[i] + bf2f(xwB[((size_t)ma * S_ + t) * G_ + nc1]);
    float gc = c10[i] + bf2f(xwB[((size_t)mb * S_ + t) * G_ + nc0]);
    float gd = c11[i] + bf2f(xwB[((size_t)mb * S_ + t) * G_ + nc1]);
    gact[(size_t)ma * G_ + nc0] = (g0t == 2) ? tanhf(ga) : 1.0f / (1.0f + __expf(-ga));
    gact[(size_t)ma * G_ + nc1] = (g1t == 2) ? tanhf(gb) : 1.0f / (1.0f + __expf(-gb));
    gact[(size_t)mb * G_ + nc0] = (g0t == 2) ? tanhf(gc) : 1.0f / (1.0f + __expf(-gc));
    gact[(size_t)mb * G_ + nc1] = (g1t == 2) ? tanhf(gd) : 1.0f / (1.0f + __expf(-gd));
  }
}

// ---------------------------------------------------------------------------
// Phase 2b (per timestep): c = f*c + i*g ; h = o*tanh(c)
// ---------------------------------------------------------------------------
__global__ void k_lstm_cell(const float* __restrict__ gact,   // (B,4H)
                            float* __restrict__ cstate,       // (B,H)
                            uint16_t* __restrict__ hidB,      // (B,S,H) bf16
                            float* __restrict__ hOut,         // (B,H) final h
                            float* __restrict__ cOut,         // (B,H) final c
                            int t) {
  int idx = blockIdx.x * blockDim.x + threadIdx.x;
  if (idx >= B_ * H_) return;
  const int b = idx >> 10;          // / H
  const int h = idx & (H_ - 1);
  const float* gb = gact + (size_t)b * G_;
  const float ig = gb[h];
  const float fg = gb[H_ + h];
  const float gg = gb[2 * H_ + h];
  const float og = gb[3 * H_ + h];
  float c = fg * cstate[idx] + ig * gg;
  float hv = og * tanhf(c);
  cstate[idx] = c;
  hidB[((size_t)b * S_ + t) * H_ + h] = f2bf(hv);
  if (t == S_ - 1) { hOut[idx] = hv; cOut[idx] = c; }
}

// ---------------------------------------------------------------------------
// Phase 3: logits = hidden_seq @ dec_w^T + dec_b  (16384 x 10000, K=1024)
//
// TDM-staged macro-tile GEMM: block (256 thr) -> 64(M) x 128(N) tile; 8 waves
// as 2x4, each wave a 2x2 grid of 16x16 accs.  Per Kc=64 chunk, wave 0 DMAs
// A panel (64x64) and B panel (128x64) into LDS via tensor_load_to_lds
// (double-buffered, s_wait_tensorcnt 2 overlap, TDM pad -> 72-elem lines).
// TDM zero-fills rows >= V on the ragged last N block.
// ---------------------------------------------------------------------------
#define DEC_MT   64
#define DEC_NT   128
#define DEC_KC   64
#define DEC_ROWP 72                       // LDS line stride in bf16 elements
#define DEC_A_ELEMS (DEC_MT * DEC_ROWP)   // 4608
#define DEC_STAGE_ELEMS ((DEC_MT + DEC_NT) * DEC_ROWP)   // 13824
#define DEC_LDS_BYTES (2 * DEC_STAGE_ELEMS * 2)          // 55296

__global__ void k_decoder(const uint16_t* __restrict__ hidB,  // (B*S,H) bf16
                          const uint16_t* __restrict__ decB,  // (V,H) bf16
                          const float* __restrict__ dec_b,    // (V)
                          float* __restrict__ out) {          // (B*S,V) f32
  extern __shared__ uint16_t smem[];   // dynamic LDS -> segment offset 0

  const int wave = threadIdx.x >> 5;
  const int lane = threadIdx.x & 31;
  const int l15  = lane & 15;
  const int aoff = (lane & 16) ? 16 : 0;
  const int boff = (lane & 16) ? 16 : 0;

  const int m0 = blockIdx.x * DEC_MT;        // 0..16320 (exact)
  const int n0 = blockIdx.y * DEC_NT;        // last block ragged vs V
  const int wm = (wave >> 2) * 32;           // 0,32
  const int wn = (wave & 3) * 32;            // 0,32,64,96

  v8f acc00 = {}, acc01 = {}, acc10 = {}, acc11 = {};

  const int nk = H_ / DEC_KC;                // 16 stages

  // Stage kc-chunk `s` into buffer s&1 (wave 0 only; EXEC-independent DMA).
  auto issue = [&](int s) {
    const uint32_t base = (uint32_t)((s & 1) * DEC_STAGE_ELEMS * 2);
    const int kc = s * DEC_KC;
    tdm_load_2d_bf16(base,
                     hidB + (size_t)m0 * H_ + kc,
                     H_, B_ * S_, H_, DEC_KC, DEC_MT);
    tdm_load_2d_bf16(base + DEC_A_ELEMS * 2,
                     decB + (size_t)n0 * H_ + kc,
                     H_, V_, H_, DEC_KC, DEC_NT);
  };
  if (threadIdx.x < 32) issue(0);

  for (int s = 0; s < nk; ++s) {
    if (threadIdx.x < 32) {
      if (s + 1 < nk) {
        issue(s + 1);                                // overlap next stage DMA
        __builtin_amdgcn_s_wait_tensorcnt((short)2); // stage s complete
      } else {
        __builtin_amdgcn_s_wait_tensorcnt((short)0);
      }
    }
    __syncthreads();                                 // stage s visible to all

    const uint16_t* sa = smem + (s & 1) * DEC_STAGE_ELEMS;
    const uint16_t* sb = sa + DEC_A_ELEMS;
#pragma unroll
    for (int k0 = 0; k0 < DEC_KC; k0 += 32) {
      v16bf a0 = load_frag_a(sa + (wm + l15) * DEC_ROWP, k0, aoff);
      v16bf a1 = load_frag_a(sa + (wm + 16 + l15) * DEC_ROWP, k0, aoff);
      v16bf b0 = load_frag_b(sb + (wn + l15) * DEC_ROWP, k0, boff);
      v16bf b1 = load_frag_b(sb + (wn + 16 + l15) * DEC_ROWP, k0, boff);
      acc00 = wmma_bf16(a0, b0, acc00);
      acc01 = wmma_bf16(a0, b1, acc01);
      acc10 = wmma_bf16(a1, b0, acc10);
      acc11 = wmma_bf16(a1, b1, acc11);
    }
    __syncthreads();   // all waves done reading before buffer is re-filled
  }

  // Write back 4 tiles; guard the ragged N edge per lane (after all WMMAs).
  const int mrow = m0 + wm + ((lane & 16) ? 8 : 0);
  const int ncol0 = n0 + wn + l15;
  const int ncol1 = ncol0 + 16;
  const float bv0 = (ncol0 < V_) ? dec_b[ncol0] : 0.0f;
  const float bv1 = (ncol1 < V_) ? dec_b[ncol1] : 0.0f;
#pragma unroll
  for (int i = 0; i < 8; ++i) {
    if (ncol0 < V_) {
      out[(size_t)(mrow + i) * V_ + ncol0]      = acc00[i] + bv0;
      out[(size_t)(mrow + 16 + i) * V_ + ncol0] = acc10[i] + bv0;
    }
    if (ncol1 < V_) {
      out[(size_t)(mrow + i) * V_ + ncol1]      = acc01[i] + bv1;
      out[(size_t)(mrow + 16 + i) * V_ + ncol1] = acc11[i] + bv1;
    }
  }
}

// ---------------------------------------------------------------------------
extern "C" void kernel_launch(void* const* d_in, const int* in_sizes, int n_in,
                              void* d_out, int out_size, void* d_ws, size_t ws_size,
                              hipStream_t stream) {
  (void)in_sizes; (void)n_in; (void)out_size; (void)ws_size;
  const int*   inp  = (const int*)d_in[0];
  const float* emb  = (const float*)d_in[1];   // (V,E)
  const float* W    = (const float*)d_in[2];   // (E,4H)
  const float* U    = (const float*)d_in[3];   // (H,4H)
  const float* bias = (const float*)d_in[4];   // (4H)
  const float* decw = (const float*)d_in[5];   // (V,H)
  const float* decb = (const float*)d_in[6];   // (V)

  float* out  = (float*)d_out;
  float* hOut = out + (size_t)B_ * S_ * V_;
  float* cOut = hOut + (size_t)B_ * H_;

  char* ws = (char*)d_ws;
  size_t off = 0;
  auto alloc = [&](size_t bytes) -> char* {
    char* p = ws + off;
    off = (off + bytes + 255) & ~(size_t)255;
    return p;
  };
  uint16_t* embB = (uint16_t*)alloc((size_t)V_ * E_ * 2);       // ~10 MB
  uint16_t* wT   = (uint16_t*)alloc((size_t)G_ * E_ * 2);       //  4 MB
  uint16_t* uT   = (uint16_t*)alloc((size_t)G_ * H_ * 2);       //  8 MB
  uint16_t* decB = (uint16_t*)alloc((size_t)V_ * H_ * 2);       // ~20 MB
  uint16_t* xwB  = (uint16_t*)alloc((size_t)B_ * S_ * G_ * 2);  // 128 MB
  uint16_t* hidB = (uint16_t*)alloc((size_t)B_ * S_ * H_ * 2);  //  32 MB
  float*    cst  = (float*)   alloc((size_t)B_ * H_ * 4);       // 0.5 MB
  float*    gact = (float*)   alloc((size_t)B_ * G_ * 4);       //  2 MB

  // One-time precision conversion / transposition of weights.
  k_cvt  <<<dim3(2048), dim3(256), 0, stream>>>(emb, embB, V_ * E_);
  k_cvt_t<<<dim3(2048), dim3(256), 0, stream>>>(W, wT, E_, G_);
  k_cvt_t<<<dim3(2048), dim3(256), 0, stream>>>(U, uT, H_, G_);
  k_cvt  <<<dim3(2048), dim3(256), 0, stream>>>(decw, decB, V_ * H_);
  k_zero <<<dim3((B_ * H_ + 255) / 256), dim3(256), 0, stream>>>(cst, B_ * H_);

  // Phase 1: gather + input GEMM for all timesteps at once (2x2 tiles).
  k_embed_gemm<<<dim3(512, 16), dim3(256), 0, stream>>>(inp, embB, wT, bias, xwB);

  // Phase 2: serial LSTM scan (stream-ordered dependency chain).
  for (int t = 0; t < S_; ++t) {
    const uint16_t* hp = (t == 0) ? nullptr : (hidB + (size_t)(t - 1) * H_);
    k_lstm_gates<<<dim3(4, 16), dim3(256), 0, stream>>>(hp, uT, xwB, gact, t);
    k_lstm_cell <<<dim3((B_ * H_ + 255) / 256), dim3(256), 0, stream>>>(
        gact, cst, hidB, hOut, cOut, t);
  }

  // Phase 3: decoder GEMM (dominant compute) with TDM-staged LDS tiles.
  k_decoder<<<dim3(B_ * S_ / DEC_MT, (V_ + DEC_NT - 1) / DEC_NT),
              dim3(256), DEC_LDS_BYTES, stream>>>(hidB, decB, decb, out);
}
